// GNN_40578851013274
// MI455X (gfx1250) — matrix operand order repaired and checked by reading, not compiled, not run on hardware
//
#include <hip/hip_runtime.h>
#include <hip/hip_bf16.h>
#include <math.h>

// Problem constants
#define BB 16
#define NN 1024
#define DD 512
#define TSTEP 3
#define BND (BB * NN * DD)   // 8,388,608 elements

typedef __bf16 bf16_t;
typedef __attribute__((ext_vector_type(16))) __bf16 v16bf;
typedef __attribute__((ext_vector_type(8)))  float  v8f;

#define WMMA_BF16(A, Bm, C) \
    __builtin_amdgcn_wmma_f32_16x16x32_bf16(false, (A), false, (Bm), (short)0, (C), false, false)

// ---------------- fragment loaders ----------------
// A fragment: 16x32 bf16 tile, row-major source, leading dim ld (elements).
// lane<16 : row = row0+lane, K = k0+0..7 and k0+16..23
// lane>=16: row = row0+lane-16, K = k0+8..15 and k0+24..31
__device__ __forceinline__ v16bf load_a_frag(const bf16_t* __restrict__ p, int ld,
                                             int row0, int k0) {
    int lane = threadIdx.x & 31;
    int r  = row0 + (lane & 15);
    int kb = k0 + ((lane >> 4) << 3);
    const bf16_t* base = p + (size_t)r * ld + kb;
    v16bf f;
    uint4* v = reinterpret_cast<uint4*>(&f);
    v[0] = *reinterpret_cast<const uint4*>(base);        // K kb..kb+7
    v[1] = *reinterpret_cast<const uint4*>(base + 16);   // K kb+16..kb+23
    return f;
}

// B fragment: 32x16 bf16 tile. Source stored "column-major" = W[col][k] with
// leading dim ld (natural [Dout,K] weight layout, or nodes_T[d][n]).
// lane<16 : col = col0+lane, K = k0+0..7 and k0+16..23 ; lane>=16: same col, K = k0+8..15 and k0+24..31
__device__ __forceinline__ v16bf load_b_frag(const bf16_t* __restrict__ p, int ld,
                                             int col0, int k0) {
    int lane = threadIdx.x & 31;
    int c  = col0 + (lane & 15);
    int kb = k0 + ((lane >> 4) << 3);
    const bf16_t* base = p + (size_t)c * ld + kb;
    v16bf f;
    uint4* v = reinterpret_cast<uint4*>(&f);
    v[0] = *reinterpret_cast<const uint4*>(base);        // K kb..kb+7
    v[1] = *reinterpret_cast<const uint4*>(base + 16);   // K kb+16..kb+23
    return f;
}

// ---------------- small utility kernels ----------------
__global__ void f32_to_bf16_kernel(const float* __restrict__ in,
                                   bf16_t* __restrict__ out, int n) {
    int i = blockIdx.x * blockDim.x + threadIdx.x;
    int stride = gridDim.x * blockDim.x;
    for (; i < n; i += stride) out[i] = (bf16_t)in[i];
}

__global__ void copy_f32_kernel(const float* __restrict__ in,
                                float* __restrict__ out, int n) {
    int i = blockIdx.x * blockDim.x + threadIdx.x;
    int stride = gridDim.x * blockDim.x;
    for (; i < n; i += stride) out[i] = in[i];
}

// nodes [B,N,D] f32 -> nodes_T [B,D,N] bf16 (LDS-tiled 32x32 transpose)
__global__ __launch_bounds__(256)
void transpose_convert_kernel(const float* __restrict__ in, bf16_t* __restrict__ outT) {
    __shared__ float tile[32][33];
    int b  = blockIdx.z;
    int n0 = blockIdx.x * 32;
    int d0 = blockIdx.y * 32;
    for (int j = threadIdx.y; j < 32; j += 8)
        tile[j][threadIdx.x] = in[((size_t)b * NN + n0 + j) * DD + d0 + threadIdx.x];
    __syncthreads();
    for (int j = threadIdx.y; j < 32; j += 8)
        outT[((size_t)b * DD + d0 + j) * NN + n0 + threadIdx.x] =
            (bf16_t)tile[threadIdx.x][j];
}

// ---------------- propagation: av = [in_m @ nodes | out_m @ nodes] ----------------
// One wave per 16(rows) x 32(cols) tile, computed for BOTH in and out matrices:
// 4 accumulators, 4 fragment loads per k-step (each fragment feeds 2 WMMAs).
__global__ __launch_bounds__(128)
void prop_kernel(const bf16_t* __restrict__ in_bf, const bf16_t* __restrict__ out_bf,
                 const bf16_t* __restrict__ nodesT, bf16_t* __restrict__ av) {
    int wave = (blockIdx.x * blockDim.x + threadIdx.x) >> 5;
    int lane = threadIdx.x & 31;
    const int MT = NN / 16, CT = DD / 32;       // 64 row tiles, 16 col super-tiles
    int b   = wave / (MT * CT);
    int rem = wave % (MT * CT);
    int m0  = (rem / CT) * 16;
    int c0  = (rem % CT) * 32;
    const bf16_t* Bp = nodesT + (size_t)b * DD * NN;

    v8f aI0 = {}, aI1 = {}, aO0 = {}, aO1 = {};
    for (int k = 0; k < NN; k += 32) {
        v16bf b0 = load_b_frag(Bp, NN, c0,      k);
        v16bf b1 = load_b_frag(Bp, NN, c0 + 16, k);
        v16bf ai = load_a_frag(in_bf,  NN, m0, k);
        v16bf ao = load_a_frag(out_bf, NN, m0, k);
        aI0 = WMMA_BF16(ai, b0, aI0);
        aI1 = WMMA_BF16(ai, b1, aI1);
        aO0 = WMMA_BF16(ao, b0, aO0);
        aO1 = WMMA_BF16(ao, b1, aO1);
    }
    int mbase = m0 + ((lane >> 4) << 3);
    int col   = c0 + (lane & 15);
    bf16_t* avb = av + (size_t)b * NN * (2 * DD);
    for (int i = 0; i < 8; ++i) {
        size_t r = (size_t)(mbase + i) * (2 * DD);
        avb[r + col]           = (bf16_t)aI0[i];
        avb[r + col + 16]      = (bf16_t)aI1[i];
        avb[r + DD + col]      = (bf16_t)aO0[i];
        avb[r + DD + col + 16] = (bf16_t)aO1[i];
    }
}

// ---------------- gates: z,r,h logits + fused sigmoid epilogue ----------------
// One wave per 32(rows) x 16(cols): 6 accumulators.
// Phase 1 (K=2D): 5 loads -> 6 WMMAs. Phase 2 (K=D, shared u3): 3 loads -> 4 WMMAs.
__global__ __launch_bounds__(128)
void gate_kernel(const bf16_t* __restrict__ av, const bf16_t* __restrict__ f_bf,
                 const bf16_t* __restrict__ w3w, const bf16_t* __restrict__ w4w,
                 const bf16_t* __restrict__ w5w, const bf16_t* __restrict__ w3u,
                 const float* __restrict__ b3w, const float* __restrict__ b3u,
                 const float* __restrict__ b4w, const float* __restrict__ b5w,
                 const float* __restrict__ fnodes,
                 float* __restrict__ zv, float* __restrict__ hvpre,
                 bf16_t* __restrict__ g) {
    int wave = (blockIdx.x * blockDim.x + threadIdx.x) >> 5;
    int lane = threadIdx.x & 31;
    const int CT = DD / 16;                     // 32 col tiles
    int r0 = (wave / CT) * 32;                  // row super-tile in [0, B*N)
    int c0 = (wave % CT) * 16;                  // output feature e

    v8f az0 = {}, ar0 = {}, ah0 = {}, az1 = {}, ar1 = {}, ah1 = {};
    for (int k = 0; k < 2 * DD; k += 32) {      // av @ {w3w,w4w,w5w}
        v16bf a0 = load_a_frag(av, 2 * DD, r0,      k);
        v16bf a1 = load_a_frag(av, 2 * DD, r0 + 16, k);
        v16bf bz = load_b_frag(w3w, 2 * DD, c0, k);
        v16bf br = load_b_frag(w4w, 2 * DD, c0, k);
        v16bf bh = load_b_frag(w5w, 2 * DD, c0, k);
        az0 = WMMA_BF16(a0, bz, az0);  az1 = WMMA_BF16(a1, bz, az1);
        ar0 = WMMA_BF16(a0, br, ar0);  ar1 = WMMA_BF16(a1, br, ar1);
        ah0 = WMMA_BF16(a0, bh, ah0);  ah1 = WMMA_BF16(a1, bh, ah1);
    }
    for (int k = 0; k < DD; k += 32) {          // shared u3 = f @ w3u (z and r)
        v16bf a0 = load_a_frag(f_bf, DD, r0,      k);
        v16bf a1 = load_a_frag(f_bf, DD, r0 + 16, k);
        v16bf bu = load_b_frag(w3u, DD, c0, k);
        az0 = WMMA_BF16(a0, bu, az0);  az1 = WMMA_BF16(a1, bu, az1);
        ar0 = WMMA_BF16(a0, bu, ar0);  ar1 = WMMA_BF16(a1, bu, ar1);
    }

    int col = c0 + (lane & 15);
    float bz = b3w[col] + b3u[col];
    float br = b4w[col] + b3u[col];
    float bh = b5w[col];
    for (int half = 0; half < 2; ++half) {
        const v8f& az = half ? az1 : az0;
        const v8f& ar = half ? ar1 : ar0;
        const v8f& ah = half ? ah1 : ah0;
        int rbase = r0 + half * 16 + ((lane >> 4) << 3);
        for (int i = 0; i < 8; ++i) {
            size_t idx = (size_t)(rbase + i) * DD + col;
            float fv = fnodes[idx];
            float z  = 1.0f / (1.0f + __expf(-(az[i] + bz)));
            float r  = 1.0f / (1.0f + __expf(-(ar[i] + br)));
            zv[idx]    = z;
            hvpre[idx] = ah[i] + bh;
            g[idx]     = (bf16_t)(r * fv);      // rv * f for w5u GEMM
        }
    }
}

// ---------------- hu = g @ w5u, fused tanh + GRU update ----------------
// One wave per 32x32 tile: 4 accumulators, 4 loads -> 4 WMMAs per k-step.
// Also emits bf16 copy of the new nodes for the next step's gate kernel.
__global__ __launch_bounds__(128)
void hu_kernel(const bf16_t* __restrict__ g, const bf16_t* __restrict__ w5u,
               const float* __restrict__ hvpre, const float* __restrict__ zv,
               const float* __restrict__ fnodes, const float* __restrict__ b5u,
               float* __restrict__ nodes_out, bf16_t* __restrict__ nodes_bf_out) {
    int wave = (blockIdx.x * blockDim.x + threadIdx.x) >> 5;
    int lane = threadIdx.x & 31;
    const int CT = DD / 32;                     // 16 col super-tiles
    int r0 = (wave / CT) * 32;
    int c0 = (wave % CT) * 32;

    v8f a00 = {}, a01 = {}, a10 = {}, a11 = {};
    for (int k = 0; k < DD; k += 32) {
        v16bf a0 = load_a_frag(g, DD, r0,      k);
        v16bf a1 = load_a_frag(g, DD, r0 + 16, k);
        v16bf b0 = load_b_frag(w5u, DD, c0,      k);
        v16bf b1 = load_b_frag(w5u, DD, c0 + 16, k);
        a00 = WMMA_BF16(a0, b0, a00);  a01 = WMMA_BF16(a0, b1, a01);
        a10 = WMMA_BF16(a1, b0, a10);  a11 = WMMA_BF16(a1, b1, a11);
    }
    int colA = c0 + (lane & 15);
    int colB = colA + 16;
    float buA = b5u[colA], buB = b5u[colB];
    for (int half = 0; half < 2; ++half) {
        const v8f& c0acc = half ? a10 : a00;
        const v8f& c1acc = half ? a11 : a01;
        int rbase = r0 + half * 16 + ((lane >> 4) << 3);
        for (int i = 0; i < 8; ++i) {
            size_t row = (size_t)(rbase + i) * DD;
            {
                size_t idx = row + colA;
                float hv = tanhf(c0acc[i] + hvpre[idx] + buA);
                float z  = zv[idx];
                float nv = (1.0f - z) * fnodes[idx] + z * hv;
                nodes_out[idx]    = nv;
                nodes_bf_out[idx] = (bf16_t)nv;
            }
            {
                size_t idx = row + colB;
                float hv = tanhf(c1acc[i] + hvpre[idx] + buB);
                float z  = zv[idx];
                float nv = (1.0f - z) * fnodes[idx] + z * hv;
                nodes_out[idx]    = nv;
                nodes_bf_out[idx] = (bf16_t)nv;
            }
        }
    }
}

// ---------------- launch ----------------
extern "C" void kernel_launch(void* const* d_in, const int* in_sizes, int n_in,
                              void* d_out, int out_size, void* d_ws, size_t ws_size,
                              hipStream_t stream) {
    const float* x    = (const float*)d_in[0];
    const float* in_m = (const float*)d_in[1];
    const float* out_m= (const float*)d_in[2];
    const float* w3w  = (const float*)d_in[3];
    const float* b3w  = (const float*)d_in[4];
    const float* w3u  = (const float*)d_in[5];
    const float* b3u  = (const float*)d_in[6];
    const float* w4w  = (const float*)d_in[7];
    const float* b4w  = (const float*)d_in[8];
    const float* w5w  = (const float*)d_in[9];
    const float* b5w  = (const float*)d_in[10];
    const float* w5u  = (const float*)d_in[11];
    const float* b5u  = (const float*)d_in[12];
    float* out = (float*)d_out;

    // workspace carve (256B aligned)
    char* ws = (char*)d_ws;
    size_t off = 0;
    auto carve = [&](size_t bytes) {
        void* p = ws + off;
        off = (off + bytes + 255) & ~(size_t)255;
        return p;
    };
    bf16_t* nodes_bf  = (bf16_t*)carve((size_t)BND * 2);              // [B,N,D] bf16
    bf16_t* nodes_bfT = (bf16_t*)carve((size_t)BND * 2);              // [B,D,N] bf16
    bf16_t* av_bf     = (bf16_t*)carve((size_t)BB * NN * 2 * DD * 2); // [B,N,2D] bf16
    bf16_t* g_bf      = (bf16_t*)carve((size_t)BND * 2);
    float*  zv        = (float*)carve((size_t)BND * 4);
    float*  hvpre     = (float*)carve((size_t)BND * 4);
    float*  nodesA    = (float*)carve((size_t)BND * 4);
    bf16_t* in_bf     = (bf16_t*)carve((size_t)NN * NN * 2);
    bf16_t* out_bf    = (bf16_t*)carve((size_t)NN * NN * 2);
    bf16_t* w3w_bf    = (bf16_t*)carve((size_t)DD * 2 * DD * 2);
    bf16_t* w4w_bf    = (bf16_t*)carve((size_t)DD * 2 * DD * 2);
    bf16_t* w5w_bf    = (bf16_t*)carve((size_t)DD * 2 * DD * 2);
    bf16_t* w3u_bf    = (bf16_t*)carve((size_t)DD * DD * 2);
    bf16_t* w5u_bf    = (bf16_t*)carve((size_t)DD * DD * 2);

    // one-time weight conversions (constant across time steps)
    f32_to_bf16_kernel<<<2048, 256, 0, stream>>>(in_m,  in_bf,  NN * NN);
    f32_to_bf16_kernel<<<2048, 256, 0, stream>>>(out_m, out_bf, NN * NN);
    f32_to_bf16_kernel<<<1024, 256, 0, stream>>>(w3w, w3w_bf, DD * 2 * DD);
    f32_to_bf16_kernel<<<1024, 256, 0, stream>>>(w4w, w4w_bf, DD * 2 * DD);
    f32_to_bf16_kernel<<<1024, 256, 0, stream>>>(w5w, w5w_bf, DD * 2 * DD);
    f32_to_bf16_kernel<<< 512, 256, 0, stream>>>(w3u, w3u_bf, DD * DD);
    f32_to_bf16_kernel<<< 512, 256, 0, stream>>>(w5u, w5u_bf, DD * DD);

    const int PROP_BLOCKS = (BB * (NN / 16) * (DD / 32)) / 4;        // 4096
    const int GATE_BLOCKS = ((BB * NN / 32) * (DD / 16)) / 4;        // 4096
    const int HU_BLOCKS   = ((BB * NN / 32) * (DD / 32)) / 4;        // 2048

    for (int s = 0; s < TSTEP; ++s) {
        const float* cur = (s == 0) ? x : nodesA;
        float* next = (s == TSTEP - 1) ? out : nodesA;

        if (s == 0)   // later steps: hu_kernel already wrote nodes_bf
            f32_to_bf16_kernel<<<8192, 256, 0, stream>>>(cur, nodes_bf, BND);
        transpose_convert_kernel<<<dim3(NN / 32, DD / 32, BB), dim3(32, 8), 0, stream>>>(
            cur, nodes_bfT);
        prop_kernel<<<PROP_BLOCKS, 128, 0, stream>>>(in_bf, out_bf, nodes_bfT, av_bf);
        gate_kernel<<<GATE_BLOCKS, 128, 0, stream>>>(av_bf, nodes_bf,
            w3w_bf, w4w_bf, w5w_bf, w3u_bf, b3w, b3u, b4w, b5w, cur,
            zv, hvpre, g_bf);
        hu_kernel<<<HU_BLOCKS, 128, 0, stream>>>(g_bf, w5u_bf, hvpre, zv, cur,
            b5u, next, nodes_bf);
    }
    // tuple output: (nodes, in_matrix)
    copy_f32_kernel<<<2048, 256, 0, stream>>>(in_m, out + (size_t)BND, NN * NN);
}
